// HAN_52183852646752
// MI455X (gfx1250) — matrix operand order repaired and checked by reading, not compiled, not run on hardware
//
#include <hip/hip_runtime.h>
#include <hip/hip_bf16.h>

typedef __attribute__((ext_vector_type(16))) __bf16 v16bf;
typedef __attribute__((ext_vector_type(8)))  float  v8f;
typedef __attribute__((ext_vector_type(4)))  int    v4i;

#define N_NODES 100000
#define IN_DIM  128
#define HID     256
#define HEADS   8
#define DH      32
#define T_TYPES 4
#define E_EDGES 400000
#define OUT_DIM 4
#define XC_DIM  384   // IN + HID

#define GEMM_WAVES   10          // waves per block = M-tiles per block (6250 = 625*10)
#define GEMM_THREADS (GEMM_WAVES * 32)
#define BLK_COLS     128         // 8 N-tiles per block
#define PROJ_LDS_STRIDE (XC_DIM + 8)   // 392 bf16 -> 784B row, 4 banks skew
#define SEM_LDS_STRIDE  (HID + 8)      // 264 bf16 -> 528B row, 4 banks skew
#define PROJ_LDS_BYTES  (BLK_COLS * PROJ_LDS_STRIDE * 2)  // 100352
#define SEM_LDS_BYTES   (BLK_COLS * SEM_LDS_STRIDE * 2)   // 67584

union AFrag { v16bf v; uint4 u[2]; };

#if defined(__has_builtin)
#if __has_builtin(__builtin_amdgcn_global_load_async_to_lds_b128)
#define HAVE_ASYNC_LDS 1
#endif
#endif
#ifndef HAVE_ASYNC_LDS
#define HAVE_ASYNC_LDS 0
#endif

typedef __attribute__((address_space(1))) v4i* gptr_v4i;
typedef __attribute__((address_space(3))) v4i* lptr_v4i;

__device__ __forceinline__ void copy16_g2l(const __bf16* g, __bf16* l) {
#if HAVE_ASYNC_LDS
  __builtin_amdgcn_global_load_async_to_lds_b128((gptr_v4i)g, (lptr_v4i)l, 0, 0);
#else
  *(uint4*)l = *(const uint4*)g;
#endif
}

__device__ __forceinline__ void fill_fence_barrier() {
#if HAVE_ASYNC_LDS
#if __has_builtin(__builtin_amdgcn_s_wait_asynccnt)
  __builtin_amdgcn_s_wait_asynccnt(0);
#else
  asm volatile("s_wait_asynccnt 0x0" ::: "memory");
#endif
#endif
  __syncthreads();
}

__device__ __forceinline__ unsigned fenc(float f) {
  unsigned u = __float_as_uint(f);
  return (u & 0x80000000u) ? ~u : (u | 0x80000000u);
}
__device__ __forceinline__ float fdec(unsigned u) {
  unsigned v = (u & 0x80000000u) ? (u & 0x7FFFFFFFu) : ~u;
  return __uint_as_float(v);
}

// ---------- prep ----------
__global__ void k_concat_bf16(const float* __restrict__ x, const int* __restrict__ nt,
                              const float* __restrict__ temb, __bf16* __restrict__ xc) {
  long total = (long)N_NODES * XC_DIM;
  for (long i = (long)blockIdx.x * blockDim.x + threadIdx.x; i < total;
       i += (long)gridDim.x * blockDim.x) {
    int n = (int)(i / XC_DIM), j = (int)(i % XC_DIM);
    float v = (j < IN_DIM) ? x[(long)n * IN_DIM + j]
                           : temb[(long)nt[n] * HID + (j - IN_DIM)];
    xc[i] = (__bf16)v;
  }
}

// W: [K, Ncols] row-major fp32 -> Wt: [Ncols, K] bf16 (column-contiguous for B frags)
__global__ void k_transpose_bf16(const float* __restrict__ W, __bf16* __restrict__ Wt,
                                 int K, int Ncols) {
  int total = K * Ncols;
  for (int i = blockIdx.x * blockDim.x + threadIdx.x; i < total;
       i += gridDim.x * blockDim.x) {
    int n = i / K, k = i % K;
    Wt[i] = (__bf16)W[(long)k * Ncols + n];
  }
}

__global__ void k_zero_u32(unsigned* __restrict__ p, long n) {
  for (long i = (long)blockIdx.x * blockDim.x + threadIdx.x; i < n;
       i += (long)gridDim.x * blockDim.x)
    p[i] = 0u;
}

// ---------- projection GEMM: xp = xc @ proj_W + b ----------
// Block: 10 waves x (8 N-tiles), B panel (128 cols x 384 K) staged in LDS via
// async-to-LDS; per-wave A fragments for all 12 K-steps held in registers.
__global__ void k_proj_wmma(const __bf16* __restrict__ xc, const __bf16* __restrict__ Wt,
                            const float* __restrict__ bias, float* __restrict__ xp) {
  extern __shared__ __bf16 smemB[];
  const int tid  = threadIdx.x;
  const int lane = tid & 31;
  const int wave = tid >> 5;
  const int half = lane >> 4;
  const int l16  = lane & 15;
  const int nbase = blockIdx.x * BLK_COLS;
  const int mtile = blockIdx.y * GEMM_WAVES + wave;

  // stage B panel: 128 rows x 48 chunks of 8 bf16 (16B)
  for (int c = tid; c < BLK_COLS * (XC_DIM / 8); c += GEMM_THREADS) {
    int row = c / (XC_DIM / 8), ch = c % (XC_DIM / 8);
    copy16_g2l(Wt + (size_t)(nbase + row) * XC_DIM + ch * 8,
               smemB + row * PROJ_LDS_STRIDE + ch * 8);
  }

  // A fragments for all K while the fill is in flight
  const __bf16* arow = xc + (size_t)(mtile * 16 + l16) * XC_DIM;
  AFrag a[XC_DIM / 32];
#pragma unroll
  for (int k = 0; k < XC_DIM / 32; ++k) {
    a[k].u[0] = *(const uint4*)(arow + k * 32 + half * 8);
    a[k].u[1] = *(const uint4*)(arow + k * 32 + 16 + half * 8);
  }

  fill_fence_barrier();

  for (int nt = 0; nt < BLK_COLS / 16; ++nt) {
    const __bf16* bcol = smemB + (nt * 16 + l16) * PROJ_LDS_STRIDE;
    v8f c = {};
#pragma unroll
    for (int k = 0; k < XC_DIM / 32; ++k) {
      AFrag b;
      b.u[0] = *(const uint4*)(bcol + k * 32 + half * 8);
      b.u[1] = *(const uint4*)(bcol + k * 32 + 16 + half * 8);
      c = __builtin_amdgcn_wmma_f32_16x16x32_bf16(false, a[k].v, false, b.v,
                                                  (short)0, c, false, false);
    }
    int col = nbase + nt * 16 + l16;
    float bv = bias[col];
#pragma unroll
    for (int r = 0; r < 8; ++r) {
      int row = mtile * 16 + half * 8 + r;
      xp[(size_t)row * HID + col] = c[r] + bv;
    }
  }
}

// ---------- per-node attention scores: asrc/adst [N,H] ----------
__global__ void k_node_att(const float* __restrict__ xp, const float* __restrict__ att_s,
                           const float* __restrict__ att_d, float* __restrict__ asrc,
                           float* __restrict__ adst) {
  int i = blockIdx.x * blockDim.x + threadIdx.x;
  if (i >= N_NODES * HEADS) return;
  int h = i & 7;
  const float4* xr = (const float4*)(xp + (long)i * DH);
  const float4* as = (const float4*)(att_s + h * DH);
  const float4* ad = (const float4*)(att_d + h * DH);
  float ss = 0.f, sd = 0.f;
#pragma unroll
  for (int j = 0; j < 8; ++j) {
    float4 v = xr[j], a = as[j], d = ad[j];
    ss += v.x * a.x + v.y * a.y + v.z * a.z + v.w * a.w;
    sd += v.x * d.x + v.y * d.y + v.z * d.z + v.w * d.w;
  }
  asrc[i] = ss;
  adst[i] = sd;
}

// ---------- edge passes ----------
__global__ void k_edge_alpha(const int* __restrict__ ei, const float* __restrict__ asrc,
                             const float* __restrict__ adst, float* __restrict__ alpha,
                             unsigned* __restrict__ amax) {
  int i = blockIdx.x * blockDim.x + threadIdx.x;
  if (i >= E_EDGES * HEADS) return;
  int e = i >> 3, h = i & 7;
  int s = ei[e], d = ei[E_EDGES + e];
  float a = asrc[s * HEADS + h] + adst[d * HEADS + h];
  a = (a > 0.f) ? a : 0.2f * a;
  alpha[i] = a;
  atomicMax(amax + d * HEADS + h, fenc(a));
}

__global__ void k_edge_exp(const int* __restrict__ ei, float* __restrict__ alpha,
                           const unsigned* __restrict__ amax, float* __restrict__ denom) {
  int i = blockIdx.x * blockDim.x + threadIdx.x;
  if (i >= E_EDGES * HEADS) return;
  int e = i >> 3, h = i & 7;
  int d = ei[E_EDGES + e];
  float m = fdec(amax[d * HEADS + h]);
  float ex = expf(alpha[i] - m);
  alpha[i] = ex;
  atomicAdd(denom + d * HEADS + h, ex);
}

__global__ void k_edge_agg(const int* __restrict__ ei, const float* __restrict__ alpha,
                           const float* __restrict__ denom, const float* __restrict__ xp,
                           float* __restrict__ zout) {
  int i = blockIdx.x * blockDim.x + threadIdx.x;
  if (i >= E_EDGES * HEADS) return;
  int e = i >> 3, h = i & 7;
  int s = ei[e], d = ei[E_EDGES + e];
  float w = alpha[i] / (denom[d * HEADS + h] + 1e-16f);
  const float* src = xp + (long)s * HID + h * DH;
  float* dst = zout + (long)d * HID + h * DH;
#pragma unroll
  for (int j = 0; j < DH; ++j) atomicAdd(dst + j, src[j] * w);
}

__global__ void k_relu(float* __restrict__ z, long n) {
  for (long i = (long)blockIdx.x * blockDim.x + threadIdx.x; i < n;
       i += (long)gridDim.x * blockDim.x)
    z[i] = fmaxf(z[i], 0.f);
}

// ---------- semantic attention: score[t] += sum(q * tanh(z_t @ kW + kb)) ----------
__global__ void k_sem_wmma(const float* __restrict__ z, const __bf16* __restrict__ kWt,
                           const float* __restrict__ kb, const float* __restrict__ q,
                           float* __restrict__ score) {
  extern __shared__ __bf16 smemB[];
  const int tid  = threadIdx.x;
  const int lane = tid & 31;
  const int wave = tid >> 5;
  const int half = lane >> 4;
  const int l16  = lane & 15;
  const int t = blockIdx.z;
  const int nbase = blockIdx.x * BLK_COLS;
  const int mtile = blockIdx.y * GEMM_WAVES + wave;

  // stage B panel: 128 rows x 32 chunks of 8 bf16
  for (int c = tid; c < BLK_COLS * (HID / 8); c += GEMM_THREADS) {
    int row = c / (HID / 8), ch = c % (HID / 8);
    copy16_g2l(kWt + (size_t)(nbase + row) * HID + ch * 8,
               smemB + row * SEM_LDS_STRIDE + ch * 8);
  }

  // A fragments (fp32 -> bf16 on the fly) for all K
  const float* zr = z + ((size_t)t * N_NODES + mtile * 16 + l16) * HID;
  AFrag a[HID / 32];
#pragma unroll
  for (int k = 0; k < HID / 32; ++k) {
    const float4* p1 = (const float4*)(zr + k * 32 + half * 8);
    const float4* p2 = (const float4*)(zr + k * 32 + 16 + half * 8);
    float4 f0 = p1[0], f1 = p1[1], f2 = p2[0], f3 = p2[1];
    a[k].v[0]  = (__bf16)f0.x; a[k].v[1]  = (__bf16)f0.y;
    a[k].v[2]  = (__bf16)f0.z; a[k].v[3]  = (__bf16)f0.w;
    a[k].v[4]  = (__bf16)f1.x; a[k].v[5]  = (__bf16)f1.y;
    a[k].v[6]  = (__bf16)f1.z; a[k].v[7]  = (__bf16)f1.w;
    a[k].v[8]  = (__bf16)f2.x; a[k].v[9]  = (__bf16)f2.y;
    a[k].v[10] = (__bf16)f2.z; a[k].v[11] = (__bf16)f2.w;
    a[k].v[12] = (__bf16)f3.x; a[k].v[13] = (__bf16)f3.y;
    a[k].v[14] = (__bf16)f3.z; a[k].v[15] = (__bf16)f3.w;
  }

  fill_fence_barrier();

  float sum = 0.f;
  for (int nt = 0; nt < BLK_COLS / 16; ++nt) {
    const __bf16* bcol = smemB + (nt * 16 + l16) * SEM_LDS_STRIDE;
    v8f c = {};
#pragma unroll
    for (int k = 0; k < HID / 32; ++k) {
      AFrag b;
      b.u[0] = *(const uint4*)(bcol + k * 32 + half * 8);
      b.u[1] = *(const uint4*)(bcol + k * 32 + 16 + half * 8);
      c = __builtin_amdgcn_wmma_f32_16x16x32_bf16(false, a[k].v, false, b.v,
                                                  (short)0, c, false, false);
    }
    int col = nbase + nt * 16 + l16;
    float qv = q[col], kbv = kb[col];
#pragma unroll
    for (int r = 0; r < 8; ++r) sum += tanhf(c[r] + kbv) * qv;
  }
  for (int off = 16; off; off >>= 1) sum += __shfl_xor(sum, off, 32);
  if (lane == 0) atomicAdd(score + t, sum);
}

__global__ void k_beta(const float* __restrict__ score, float* __restrict__ beta) {
  if (threadIdx.x == 0 && blockIdx.x == 0) {
    float s[T_TYPES], mx = -1e30f;
    for (int t = 0; t < T_TYPES; ++t) { s[t] = score[t] / (float)N_NODES; mx = fmaxf(mx, s[t]); }
    float den = 0.f;
    for (int t = 0; t < T_TYPES; ++t) { s[t] = expf(s[t] - mx); den += s[t]; }
    for (int t = 0; t < T_TYPES; ++t) beta[t] = s[t] / den;
  }
}

// ---------- fuse + relu + final linear ----------
__global__ void k_final(const float* __restrict__ z, const float* __restrict__ beta,
                        const float* __restrict__ linW, const float* __restrict__ linb,
                        float* __restrict__ out) {
  int lane = threadIdx.x & 31;
  int wave = threadIdx.x >> 5;
  long node = (long)blockIdx.x * (blockDim.x >> 5) + wave;
  if (node >= N_NODES) return;
  float b0 = beta[0], b1 = beta[1], b2 = beta[2], b3 = beta[3];
  const long stride = (long)N_NODES * HID;
  float acc0 = 0.f, acc1 = 0.f, acc2 = 0.f, acc3 = 0.f;
#pragma unroll
  for (int j = 0; j < 8; ++j) {
    int h = lane + 32 * j;
    long idx = node * HID + h;
    float f = b0 * z[idx] + b1 * z[idx + stride] + b2 * z[idx + 2 * stride] +
              b3 * z[idx + 3 * stride];
    f = fmaxf(f, 0.f);
    float4 w = *(const float4*)(linW + h * OUT_DIM);
    acc0 += f * w.x; acc1 += f * w.y; acc2 += f * w.z; acc3 += f * w.w;
  }
  for (int off = 16; off; off >>= 1) {
    acc0 += __shfl_xor(acc0, off, 32);
    acc1 += __shfl_xor(acc1, off, 32);
    acc2 += __shfl_xor(acc2, off, 32);
    acc3 += __shfl_xor(acc3, off, 32);
  }
  if (lane == 0) {
    out[node * OUT_DIM + 0] = acc0 + linb[0];
    out[node * OUT_DIM + 1] = acc1 + linb[1];
    out[node * OUT_DIM + 2] = acc2 + linb[2];
    out[node * OUT_DIM + 3] = acc3 + linb[3];
  }
}

extern "C" void kernel_launch(void* const* d_in, const int* in_sizes, int n_in,
                              void* d_out, int out_size, void* d_ws, size_t ws_size,
                              hipStream_t stream) {
  const float* x          = (const float*)d_in[0];
  const int*   node_types = (const int*)d_in[1];
  const int*   ei[T_TYPES] = {(const int*)d_in[2], (const int*)d_in[3],
                              (const int*)d_in[4], (const int*)d_in[5]};
  const float* type_emb = (const float*)d_in[6];
  const float* proj_W   = (const float*)d_in[7];
  const float* proj_b   = (const float*)d_in[8];
  const float* att_src  = (const float*)d_in[9];
  const float* att_dst  = (const float*)d_in[10];
  const float* q        = (const float*)d_in[11];
  const float* kW       = (const float*)d_in[12];
  const float* kb       = (const float*)d_in[13];
  const float* lin_W    = (const float*)d_in[14];
  const float* lin_b    = (const float*)d_in[15];
  float* out = (float*)d_out;

  char* ws = (char*)d_ws;
  size_t cur = 0;
  auto carve = [&](size_t bytes) -> char* {
    char* p = ws + cur;
    cur = (cur + bytes + 255) & ~(size_t)255;
    return p;
  };
  __bf16*   xc_bf  = (__bf16*)carve((size_t)N_NODES * XC_DIM * 2);
  __bf16*   Wt_bf  = (__bf16*)carve((size_t)XC_DIM * HID * 2);
  __bf16*   kWt_bf = (__bf16*)carve((size_t)HID * HID * 2);
  float*    xp     = (float*)carve((size_t)N_NODES * HID * 4);
  float*    asrc   = (float*)carve((size_t)N_NODES * HEADS * 4);
  float*    adst   = (float*)carve((size_t)N_NODES * HEADS * 4);
  unsigned* amax   = (unsigned*)carve((size_t)N_NODES * HEADS * 4);
  float*    denom  = (float*)carve((size_t)N_NODES * HEADS * 4);
  float*    alpha  = (float*)carve((size_t)E_EDGES * HEADS * 4);
  float*    z      = (float*)carve((size_t)T_TYPES * N_NODES * HID * 4);
  float*    score  = (float*)carve(T_TYPES * 4);
  float*    beta   = (float*)carve(T_TYPES * 4);

  const int TB = 256;

  k_concat_bf16<<<8192, TB, 0, stream>>>(x, node_types, type_emb, xc_bf);
  k_transpose_bf16<<<(XC_DIM * HID + TB - 1) / TB, TB, 0, stream>>>(proj_W, Wt_bf, XC_DIM, HID);
  k_transpose_bf16<<<(HID * HID + TB - 1) / TB, TB, 0, stream>>>(kW, kWt_bf, HID, HID);
  k_zero_u32<<<8192, TB, 0, stream>>>((unsigned*)z, (long)T_TYPES * N_NODES * HID);
  k_zero_u32<<<1, 32, 0, stream>>>((unsigned*)score, T_TYPES);

  // projection GEMM: 2 column-blocks x 625 row-blocks, 10 waves each
  {
    dim3 grid(HID / BLK_COLS, (N_NODES / 16) / GEMM_WAVES);
    k_proj_wmma<<<grid, GEMM_THREADS, PROJ_LDS_BYTES, stream>>>(xc_bf, Wt_bf, proj_b, xp);
  }

  const int nh_blocks = (N_NODES * HEADS + TB - 1) / TB;
  const int eh_blocks = (E_EDGES * HEADS + TB - 1) / TB;
  for (int t = 0; t < T_TYPES; ++t) {
    k_node_att<<<nh_blocks, TB, 0, stream>>>(xp, att_src + t * HEADS * DH,
                                             att_dst + t * HEADS * DH, asrc, adst);
    k_zero_u32<<<nh_blocks, TB, 0, stream>>>(amax, (long)N_NODES * HEADS);
    k_zero_u32<<<nh_blocks, TB, 0, stream>>>((unsigned*)denom, (long)N_NODES * HEADS);
    k_edge_alpha<<<eh_blocks, TB, 0, stream>>>(ei[t], asrc, adst, alpha, amax);
    k_edge_exp<<<eh_blocks, TB, 0, stream>>>(ei[t], alpha, amax, denom);
    k_edge_agg<<<eh_blocks, TB, 0, stream>>>(ei[t], alpha, denom, xp,
                                             z + (size_t)t * N_NODES * HID);
  }

  k_relu<<<8192, TB, 0, stream>>>(z, (long)T_TYPES * N_NODES * HID);

  // semantic GEMM: 2 column-blocks x 625 row-blocks x 4 types
  {
    dim3 grid(HID / BLK_COLS, (N_NODES / 16) / GEMM_WAVES, T_TYPES);
    k_sem_wmma<<<grid, GEMM_THREADS, SEM_LDS_BYTES, stream>>>(z, kWt_bf, kb, q, score);
  }
  k_beta<<<1, 32, 0, stream>>>(score, beta);

  k_final<<<(N_NODES + 7) / 8, 256, 0, stream>>>(z, beta, lin_W, lin_b, out);
}
